// InteractionBlock_48627619725655
// MI455X (gfx1250) — matrix operand order repaired and checked by reading, not compiled, not run on hardware
//
#include <hip/hip_runtime.h>
#include <hip/hip_bf16.h>

// ---------------------------------------------------------------------------
// PhysNet InteractionBlock on gfx1250 (MI455X)
//   - wave32, v_wmma_f32_16x16x32_bf16 (fp32 accum) for all GEMMs
//   - global_load_async_to_lds_b128 (+ s_wait_asynccnt) for weight staging
//   - run-length compressed global_atomic_add_f32 scatter-sum (idx_i sorted)
// ---------------------------------------------------------------------------
// d_in order (jax pytree: dict keys sorted, lists in order):
//   0 features(50000x128) 1 descriptors(1.6Mx64) 2 idx_i 3 idx_j
//   4 W_r2a(64x128)
//   5 dense_i.W  6 dense_i.b  7 dense_j.W  8 dense_j.b
//   9 dense_out.W 10 dense_out.b
//   11..18 res_feat[0..1].{l1.W,l1.b,l2.W,l2.b}
//   19..30 res_ij[0..2].{l1.W,l1.b,l2.W,l2.b}
//   31 scaling(128)
// ---------------------------------------------------------------------------

typedef __bf16 bf16_t;
typedef __attribute__((ext_vector_type(16))) __bf16 v16bf;
typedef __attribute__((ext_vector_type(8)))  float  v8f;

#define N_ATOMS   50000
#define N_PAIRS   1600000
#define TILES_N   3125     // 50000/16
#define TILES_P   100000   // 1600000/16
#define FDIM      128

__device__ __forceinline__ float sspf(float x) {
  // shifted softplus: softplus(x) - log(2), numerically stable
  return fmaxf(x, 0.0f) + log1pf(__expf(-fabsf(x))) - 0.6931471805599453f;
}

// ---------------------------------------------------------------------------
// Async stage of packed weights into LDS: global_load_async_to_lds_b128.
// Each thread copies disjoint 16B chunks; generic pointer to __shared__ has
// the LDS byte offset in its low 32 bits (LDS aperture truncation rule).
// ---------------------------------------------------------------------------
__device__ __forceinline__ void stage_weights(const bf16_t* __restrict__ gsrc,
                                              bf16_t* lds, int totalElems) {
  int t = threadIdx.x;
  for (int i = t * 8; i < totalElems; i += 256 * 8) {
    unsigned int       laddr = (unsigned int)(unsigned long long)(lds + i);
    unsigned long long gaddr = (unsigned long long)(gsrc + i);
    asm volatile("global_load_async_to_lds_b128 %0, %1, off"
                 :: "v"(laddr), "v"(gaddr) : "memory");
  }
  asm volatile("s_wait_asynccnt 0x0" ::: "memory");
  __syncthreads();
}

// Load one 16x32 bf16 A-fragment for wmma_f32_16x16x32_bf16.
// ISA layout: lane L -> row = L%16, half = L/16;
//   elements e<8  : k = k0 + half*8 + e          (contiguous 8 floats)
//   elements e>=8 : k = k0 + 16 + half*8 + (e-8) (contiguous 8 floats)
template <bool PREACT>
__device__ __forceinline__ v16bf load_a_frag(const float* __restrict__ rowPtr,
                                             int k0, int half) {
  const float4* p1 = (const float4*)(rowPtr + k0 + half * 8);
  const float4* p2 = (const float4*)(rowPtr + k0 + 16 + half * 8);
  float4 c0 = p1[0], c1 = p1[1];
  float4 c2 = p2[0], c3 = p2[1];
  float f[16] = {c0.x, c0.y, c0.z, c0.w, c1.x, c1.y, c1.z, c1.w,
                 c2.x, c2.y, c2.z, c2.w, c3.x, c3.y, c3.z, c3.w};
  v16bf r;
#pragma unroll
  for (int e = 0; e < 16; ++e) {
    float v = PREACT ? sspf(f[e]) : f[e];
    r[e] = (bf16_t)v;
  }
  return r;
}

// ---------------------------------------------------------------------------
// Pack a (Kdim x 128) fp32 row-major weight into bf16 B-fragment order:
// linear idx = ((ks*8 + nt)*32 + lane)*16 + e ;  n = nt*16 + lane%16 ;
// k = ks*32 + (lane/16)*8 + (e%8) + (e/8)*16   (same k order as the A fragment)
// ---------------------------------------------------------------------------
__global__ void pack_weight_kernel(const float* __restrict__ W,
                                   bf16_t* __restrict__ dst, int Kdim) {
  int idx = blockIdx.x * 256 + threadIdx.x;
  int total = Kdim * FDIM;
  if (idx >= total) return;
  int e    = idx & 15;
  int lane = (idx >> 4) & 31;
  int frag = idx >> 9;
  int nt   = frag & 7;
  int ks   = frag >> 3;
  int n = nt * 16 + (lane & 15);
  int k = ks * 32 + (lane >> 4) * 8 + (e & 7) + (e >> 3) * 16;
  dst[idx] = (bf16_t)W[k * FDIM + n];
}

// elementwise x = ssp(features), vectorized by 4
__global__ void ssp4_kernel(const float4* __restrict__ in,
                            float4* __restrict__ out, int n4) {
  int i = blockIdx.x * 256 + threadIdx.x;
  if (i >= n4) return;
  float4 v = in[i];
  v.x = sspf(v.x); v.y = sspf(v.y); v.z = sspf(v.z); v.w = sspf(v.w);
  out[i] = v;
}

// ---------------------------------------------------------------------------
// Generic (N x 128) @ (KDIM x 128) GEMM, bf16 WMMA, fp32 accum.
// Block = 256 threads = 8 waves; each wave computes one 16-row x 128-col stripe.
// out[r,c] = post(A'@W + b)[r,c] + (SCALE? s[c]:1) * (RESID? resid[r,c]:0)
// All behavior flags are compile-time -> no per-iteration SALU branching.
// ---------------------------------------------------------------------------
template <int KDIM, bool PREACT, bool POSTACT, bool RESID, bool SCALE>
__global__ __launch_bounds__(256)
void gemm128_kernel(const float* __restrict__ A, const bf16_t* __restrict__ Wp,
                    const float* __restrict__ bias,
                    const float* __restrict__ resid,
                    const float* __restrict__ scalevec,
                    float* __restrict__ out, int nTilesM) {
  __shared__ bf16_t ldsW[KDIM * FDIM];
  stage_weights(Wp, ldsW, KDIM * FDIM);

  int t = threadIdx.x;
  int wave = t >> 5, lane = t & 31;
  int tileM = blockIdx.x * 8 + wave;
  if (tileM >= nTilesM) return;      // uniform per wave -> EXEC all-1 for WMMA

  int half = lane >> 4, l16 = lane & 15;
  const float* rowPtr = A + (size_t)(tileM * 16 + l16) * FDIM;

  v8f acc[8];
#pragma unroll
  for (int nt = 0; nt < 8; ++nt) acc[nt] = (v8f){0.f,0.f,0.f,0.f,0.f,0.f,0.f,0.f};

#pragma unroll
  for (int ks = 0; ks < (KDIM >> 5); ++ks) {
    v16bf a = load_a_frag<PREACT>(rowPtr, ks * 32, half);
#pragma unroll
    for (int nt = 0; nt < 8; ++nt) {
      v16bf b = *(const v16bf*)&ldsW[((ks * 8 + nt) * 32 + lane) * 16];
      acc[nt] = __builtin_amdgcn_wmma_f32_16x16x32_bf16(
          false, a, false, b, (short)0, acc[nt], false, false);
    }
  }

  // D layout: element v of v8f -> row M = v + 8*half, col N = l16 (within tile)
  int row0 = tileM * 16 + 8 * half;
#pragma unroll
  for (int nt = 0; nt < 8; ++nt) {
    int c = nt * 16 + l16;
    float bv = bias[c];
    float sv = SCALE ? scalevec[c] : 1.0f;
#pragma unroll
    for (int v = 0; v < 8; ++v) {
      float r = acc[nt][v] + bv;
      if (POSTACT) r = sspf(r);
      if (RESID) r += sv * resid[(size_t)(row0 + v) * FDIM + c];
      out[(size_t)(row0 + v) * FDIM + c] = r;
    }
  }
}

// ---------------------------------------------------------------------------
// Fused pair kernel:
//   g = descriptors @ W_r2a   (16-pair tile per wave, bf16 WMMA, fp32 accum)
//   msg[idx_i[p], :] += g[p, :] * xjf[idx_j[p], :]
// idx_i is sorted -> run-length accumulate in registers, flush with native
// float atomics only on segment change (~8x fewer atomics).
// msg must be pre-initialized with xi (done by the dense_i GEMM).
// ---------------------------------------------------------------------------
__global__ __launch_bounds__(256)
void pair_kernel(const float* __restrict__ desc, const bf16_t* __restrict__ Wp,
                 const int* __restrict__ idx_i, const int* __restrict__ idx_j,
                 const float* __restrict__ xjf, float* __restrict__ msg,
                 int nTilesP) {
  __shared__ bf16_t ldsW[64 * FDIM];  // 16 KB (packed W_r2a)
  stage_weights(Wp, ldsW, 64 * FDIM);

  int t = threadIdx.x;
  int wave = t >> 5, lane = t & 31;
  int tileP = blockIdx.x * 8 + wave;
  if (tileP >= nTilesP) return;

  int half = lane >> 4, l16 = lane & 15;
  const float* rowPtr = desc + (size_t)(tileP * 16 + l16) * 64;

  v8f acc[8];
#pragma unroll
  for (int nt = 0; nt < 8; ++nt) acc[nt] = (v8f){0.f,0.f,0.f,0.f,0.f,0.f,0.f,0.f};

#pragma unroll
  for (int ks = 0; ks < 2; ++ks) {             // K = 64 -> 2 steps of 32
    v16bf a = load_a_frag<false>(rowPtr, ks * 32, half);
#pragma unroll
    for (int nt = 0; nt < 8; ++nt) {
      v16bf b = *(const v16bf*)&ldsW[((ks * 8 + nt) * 32 + lane) * 16];
      acc[nt] = __builtin_amdgcn_wmma_f32_16x16x32_bf16(
          false, a, false, b, (short)0, acc[nt], false, false);
    }
  }

  // rows handled by this half-wave: pairs pbase..pbase+7 (consecutive, sorted i)
  int pbase = tileP * 16 + 8 * half;
  int cur = idx_i[pbase];
  float carry[8];
#pragma unroll
  for (int nt = 0; nt < 8; ++nt) carry[nt] = 0.0f;

#pragma unroll
  for (int v = 0; v < 8; ++v) {
    int p = pbase + v;
    int ai = idx_i[p];
    int aj = idx_j[p];
    const float* xr = xjf + (size_t)aj * FDIM;
    if (ai != cur) {                 // flush previous segment
      float* mr = msg + (size_t)cur * FDIM;
#pragma unroll
      for (int nt = 0; nt < 8; ++nt) {
        unsafeAtomicAdd(&mr[nt * 16 + l16], carry[nt]);
        carry[nt] = 0.0f;
      }
      cur = ai;
    }
#pragma unroll
    for (int nt = 0; nt < 8; ++nt)
      carry[nt] += acc[nt][v] * xr[nt * 16 + l16];
  }
  {
    float* mr = msg + (size_t)cur * FDIM;
#pragma unroll
    for (int nt = 0; nt < 8; ++nt)
      unsafeAtomicAdd(&mr[nt * 16 + l16], carry[nt]);
  }
}

// ---------------------------------------------------------------------------
extern "C" void kernel_launch(void* const* d_in, const int* in_sizes, int n_in,
                              void* d_out, int out_size, void* d_ws, size_t ws_size,
                              hipStream_t stream) {
  (void)in_sizes; (void)n_in; (void)out_size; (void)ws_size;

  const float* features    = (const float*)d_in[0];
  const float* descriptors = (const float*)d_in[1];
  const int*   idx_i       = (const int*)d_in[2];
  const int*   idx_j       = (const int*)d_in[3];
  const float* W_r2a       = (const float*)d_in[4];
  const float* scaling     = (const float*)d_in[31];
  float* out = (float*)d_out;

  // scratch layout
  const size_t BYTES_NF = (size_t)N_ATOMS * FDIM * sizeof(float);  // 25.6 MB
  char* ws = (char*)d_ws;
  float*  X    = (float*)(ws + 0);                 // x = ssp(features)
  float*  B1   = (float*)(ws + BYTES_NF);          // xj_feat, then residual temp h
  float*  B2   = (float*)(ws + 2 * BYTES_NF);      // message
  bf16_t* Wr2a = (bf16_t*)(ws + 3 * BYTES_NF);     // packed W_r2a (16 KB)
  bf16_t* Wd   = (bf16_t*)(ws + 3 * BYTES_NF + 64 * FDIM * sizeof(bf16_t)); // 13x32KB

  // packed dense-weight slot order, with their d_in indices (W, b):
  // 0 dense_i, 1 dense_j, 2..7 res_ij[0..2].{l1,l2}, 8 dense_out, 9..12 res_feat
  const int wIdx[13] = {5, 7, 19, 21, 23, 25, 27, 29, 9, 11, 13, 15, 17};
  const int bIdx[13] = {6, 8, 20, 22, 24, 26, 28, 30, 10, 12, 14, 16, 18};

  // 1) prepack weights to bf16 fragment order
  pack_weight_kernel<<<(64 * FDIM + 255) / 256, 256, 0, stream>>>(W_r2a, Wr2a, 64);
  for (int m = 0; m < 13; ++m)
    pack_weight_kernel<<<(128 * FDIM + 255) / 256, 256, 0, stream>>>(
        (const float*)d_in[wIdx[m]], Wd + (size_t)m * 128 * FDIM, 128);

  // 2) x = ssp(features)
  int n4 = N_ATOMS * FDIM / 4;
  ssp4_kernel<<<(n4 + 255) / 256, 256, 0, stream>>>((const float4*)features, (float4*)X, n4);

  const int GB = (TILES_N + 7) / 8;  // 391 blocks for atom GEMMs
  #define WSLOT(m) (Wd + (size_t)(m) * 128 * FDIM)
  #define BIAS(m)  ((const float*)d_in[bIdx[m]])

  // 3) message = xi = ssp(x @ Wi + bi)   (written straight into B2)
  gemm128_kernel<128, false, true, false, false><<<GB, 256, 0, stream>>>(
      X, WSLOT(0), BIAS(0), nullptr, nullptr, B2, TILES_N);
  // 4) xj_feat = ssp(x @ Wj + bj)
  gemm128_kernel<128, false, true, false, false><<<GB, 256, 0, stream>>>(
      X, WSLOT(1), BIAS(1), nullptr, nullptr, B1, TILES_N);

  // 5) fused pair GEMM + gather + compressed atomic scatter-sum into message
  pair_kernel<<<TILES_P / 8, 256, 0, stream>>>(descriptors, Wr2a, idx_i, idx_j,
                                               B1, B2, TILES_P);

  // 6) res_ij x3: h = ssp(ssp(m)@W1+b1); m = m + (h@W2+b2)
  for (int tblk = 0; tblk < 3; ++tblk) {
    gemm128_kernel<128, true, true, false, false><<<GB, 256, 0, stream>>>(
        B2, WSLOT(2 + 2 * tblk), BIAS(2 + 2 * tblk), nullptr, nullptr, B1, TILES_N);
    gemm128_kernel<128, false, false, true, false><<<GB, 256, 0, stream>>>(
        B1, WSLOT(3 + 2 * tblk), BIAS(3 + 2 * tblk), B2, nullptr, B2, TILES_N);
  }

  // 7) out = scaling * x + ssp(m) @ Wout + bout
  gemm128_kernel<128, true, false, true, true><<<GB, 256, 0, stream>>>(
      B2, WSLOT(8), BIAS(8), X, scaling, out, TILES_N);

  // 8) res_feat x2 (in place on out)
  for (int tblk = 0; tblk < 2; ++tblk) {
    gemm128_kernel<128, true, true, false, false><<<GB, 256, 0, stream>>>(
        out, WSLOT(9 + 2 * tblk), BIAS(9 + 2 * tblk), nullptr, nullptr, B1, TILES_N);
    gemm128_kernel<128, false, false, true, false><<<GB, 256, 0, stream>>>(
        B1, WSLOT(10 + 2 * tblk), BIAS(10 + 2 * tblk), out, nullptr, out, TILES_N);
  }
  #undef WSLOT
  #undef BIAS
}
